// Model_20418274525858
// MI455X (gfx1250) — matrix operand order repaired and checked by reading, not compiled, not run on hardware
//
#include <hip/hip_runtime.h>

// ---------------------------------------------------------------------------
// Types for CDNA5 WMMA (wave32)
// ---------------------------------------------------------------------------
typedef __bf16 bf16_t;
typedef __attribute__((ext_vector_type(16))) __bf16 v16bf;
typedef __attribute__((ext_vector_type(8)))  float  v8f;
typedef __attribute__((ext_vector_type(4)))  unsigned int u32x4;

union Frag { u32x4 u[2]; v16bf v; };

static constexpr int N_  = 50000;   // nodes
static constexpr int R_  = 8;       // relations
static constexpr int E_  = 200000;  // edges per relation
static constexpr int ED_ = 400000;  // decoder edges

__device__ __forceinline__ void atomAddF(float* p, float v) {
  __hip_atomic_fetch_add(p, v, __ATOMIC_RELAXED, __HIP_MEMORY_SCOPE_AGENT);
}

// ---------------------------------------------------------------------------
// Utility kernels
// ---------------------------------------------------------------------------
__global__ void zero_f32(float* __restrict__ p, long n) {
  long t = (long)blockIdx.x * 256 + threadIdx.x;
  if (t < n) p[t] = 0.0f;
}

// degree counting: idx is [R,E] flat; deg is [R,N] flat
__global__ void count_deg(const int* __restrict__ idx, float* __restrict__ deg,
                          long total, int nNodes, int ePer) {
  long t = (long)blockIdx.x * 256 + threadIdx.x;
  if (t >= total) return;
  int r = (int)(t / ePer);
  int node = idx[t];
  atomAddF(deg + (long)r * nNodes + node, 1.0f);
}

// deg -> (deg>0) ? rsqrt(max(deg,1)) : 0
__global__ void norm_finalize(float* __restrict__ p, long n) {
  long t = (long)blockIdx.x * 256 + threadIdx.x;
  if (t >= n) return;
  float v = p[t];
  p[t] = (v > 0.0f) ? rsqrtf(v < 1.0f ? 1.0f : v) : 0.0f;
}

// W [batch][K][Nc] f32 -> Wt [batch][Nc][Kp] bf16 (zero pad K..Kp)
__global__ void transpose_pad_bf16(const float* __restrict__ W, bf16_t* __restrict__ Wt,
                                   int K, int Nc, int Kp, long total) {
  long t = (long)blockIdx.x * 256 + threadIdx.x;
  if (t >= total) return;
  int k = (int)(t % Kp);
  long t2 = t / Kp;
  int c = (int)(t2 % Nc);
  int b = (int)(t2 / Nc);
  float v = (k < K) ? W[((long)b * K + k) * Nc + c] : 0.0f;
  Wt[t] = (bf16_t)v;
}

// x [N,K] f32 -> xb [N,Kp] bf16 (zero pad)
__global__ void convert_pad_bf16(const float* __restrict__ x, bf16_t* __restrict__ xb,
                                 int K, int Kp, long total) {
  long t = (long)blockIdx.x * 256 + threadIdx.x;
  if (t >= total) return;
  int k = (int)(t % Kp);
  long i = t / Kp;
  xb[t] = (bf16_t)((k < K) ? x[i * K + k] : 0.0f);
}

// acc[N,dout] + sum_r b[r][dout] -> relu -> bf16
__global__ void finalize_relu_bf16(const float* __restrict__ acc, const float* __restrict__ b,
                                   bf16_t* __restrict__ out, long n, int dout) {
  long t = (long)blockIdx.x * 256 + threadIdx.x;
  if (t >= n) return;
  int j = (int)(t % dout);
  float s = acc[t];
#pragma unroll
  for (int r = 0; r < R_; ++r) s += b[r * dout + j];
  out[t] = (bf16_t)(s > 0.0f ? s : 0.0f);
}

// acc[N,dout] + sum_r b -> bf16 written into feat[:, coff:coff+dout]
__global__ void finalize_feat_bf16(const float* __restrict__ acc, const float* __restrict__ b,
                                   bf16_t* __restrict__ feat, long n, int dout,
                                   int featld, int coff) {
  long t = (long)blockIdx.x * 256 + threadIdx.x;
  if (t >= n) return;
  long i = t / dout;
  int  j = (int)(t % dout);
  float s = acc[t];
#pragma unroll
  for (int r = 0; r < R_; ++r) s += b[r * dout + j];
  feat[i * featld + coff + j] = (bf16_t)s;
}

// ---------------------------------------------------------------------------
// bf16 WMMA GEMM:  C[M,Nc] (f32) = A[M,Kp](bf16) * Bt[Nc,Kp](bf16)^T
// Block tile 64x64, 8 waves, each wave 16 rows x 32 cols (two 16x16 frags).
// ---------------------------------------------------------------------------
__global__ void __launch_bounds__(256) gemm_bf16_wmma(
    const bf16_t* __restrict__ A, const bf16_t* __restrict__ Bt,
    float* __restrict__ C, int M, int Kp, int Nc) {
  int w = threadIdx.x >> 5, lane = threadIdx.x & 31;
  int wr = w & 3, wc = w >> 2;
  int m0 = blockIdx.x * 64 + wr * 16;
  int n0 = blockIdx.y * 64 + wc * 32;

  int row = m0 + (lane & 15);
  if (row >= M) row = M - 1;              // clamp loads; stores are guarded
  int khalf = (lane < 16) ? 0 : 8;        // A fragment K split (ISA 16-bit A layout)
  int kb    = (lane < 16) ? 0 : 8;        // B fragment K split (same as A layout)

  const bf16_t* Arow = A  + (long)row * Kp;
  const bf16_t* B0   = Bt + (long)(n0 + (lane & 15)) * Kp;
  const bf16_t* B1   = B0 + (long)16 * Kp;

  v8f acc0 = {}; v8f acc1 = {};
  for (int kk = 0; kk < Kp; kk += 32) {
    Frag a, b0, b1;
    a.u[0]  = *(const u32x4*)(Arow + kk + khalf);
    a.u[1]  = *(const u32x4*)(Arow + kk + khalf + 16);
    b0.u[0] = *(const u32x4*)(B0 + kk + kb);
    b0.u[1] = *(const u32x4*)(B0 + kk + kb + 16);
    b1.u[0] = *(const u32x4*)(B1 + kk + kb);
    b1.u[1] = *(const u32x4*)(B1 + kk + kb + 16);
    acc0 = __builtin_amdgcn_wmma_f32_16x16x32_bf16(false, a.v, false, b0.v,
                                                   (short)0, acc0, false, false);
    acc1 = __builtin_amdgcn_wmma_f32_16x16x32_bf16(false, a.v, false, b1.v,
                                                   (short)0, acc1, false, false);
  }

  int rbase = m0 + ((lane >= 16) ? 8 : 0);
  int cn0 = n0 + (lane & 15);
#pragma unroll
  for (int r = 0; r < 8; ++r) {
    int rr = rbase + r;
    if (rr < M) {
      C[(long)rr * Nc + cn0]      = acc0[r];
      C[(long)rr * Nc + cn0 + 16] = acc1[r];
    }
  }
}

// ---------------------------------------------------------------------------
// Edge scatter with fused norms: acc[dst] += m[src] * ns[src]*nd[dst]
// One wave per edge; lane handles DOUT/128 float4 chunks, 128 floats apart
// (L2-resident tmp/acc: runs at L2 atomic rate).
// ---------------------------------------------------------------------------
template <int DOUT>
__global__ void __launch_bounds__(256) scatter_edges(
    const int* __restrict__ src, const int* __restrict__ dst,
    const float* __restrict__ ns, const float* __restrict__ nd,
    const float* __restrict__ m, float* __restrict__ acc, int nE) {
  int w = threadIdx.x >> 5, lane = threadIdx.x & 31;
  int e = blockIdx.x * 8 + w;
  if (e >= nE) return;
  int s = src[e], d = dst[e];
  float c = ns[s] * nd[d];
  const float* mr = m + (long)s * DOUT;
  float* ar = acc + (long)d * DOUT;
#pragma unroll
  for (int chunk = 0; chunk < DOUT / 128; ++chunk) {
    int j = chunk * 128 + lane * 4;
    float4 v = *(const float4*)(mr + j);
    atomAddF(ar + j + 0, v.x * c);
    atomAddF(ar + j + 1, v.y * c);
    atomAddF(ar + j + 2, v.z * c);
    atomAddF(ar + j + 3, v.w * c);
  }
}

// ---------------------------------------------------------------------------
// Fused decoder: per block 32 edges.
//   hidden = [feat[dsrc] , feat[ddst]] (512) @ Wp1 + bp1   (WMMA, bf16)
//   score  = hidden @ Wp2 + bp2                            (VALU from LDS)
// 8 waves: 2 row tiles (16) x 4 col tiles (64 of 256).
// ---------------------------------------------------------------------------
__global__ void __launch_bounds__(256) decoder_kernel(
    const int* __restrict__ dsrc, const int* __restrict__ ddst,
    const bf16_t* __restrict__ feat, const bf16_t* __restrict__ Wp1t,
    const float* __restrict__ bp1, const float* __restrict__ Wp2,
    const float* __restrict__ bp2, float* __restrict__ out) {
  __shared__ float hid[32 * 260];          // padded stride to kill bank conflicts
  __shared__ float wp2s[256 * 8];
  __shared__ float bp2s[8];
  __shared__ int sidx[32], didx[32];
  int tid = threadIdx.x;
  long e0 = (long)blockIdx.x * 32;
  if (tid < 32) { sidx[tid] = dsrc[e0 + tid]; didx[tid] = ddst[e0 + tid]; }
#pragma unroll
  for (int i = 0; i < 8; ++i) wp2s[i * 256 + tid] = Wp2[i * 256 + tid];
  if (tid < 8) bp2s[tid] = bp2[tid];
  __syncthreads();

  int w = tid >> 5, lane = tid & 31;
  int wr = w & 1, wc = w >> 1;
  int lr = wr * 16 + (lane & 15);
  int khalf = (lane < 16) ? 0 : 8;
  int kb    = (lane < 16) ? 0 : 8;
  int sN = sidx[lr], dN = didx[lr];

  const bf16_t* Brow[4];
#pragma unroll
  for (int s = 0; s < 4; ++s)
    Brow[s] = Wp1t + (long)(wc * 64 + s * 16 + (lane & 15)) * 512;

  v8f acc[4] = {{}, {}, {}, {}};
#pragma unroll
  for (int kk = 0; kk < 512; kk += 32) {
    int node = (kk < 256) ? sN : dN;       // concat boundary aligns to 32-wide K step
    int koff = (kk + khalf) & 255;
    const bf16_t* Arow = feat + (long)node * 256;
    Frag a;
    a.u[0] = *(const u32x4*)(Arow + koff);
    a.u[1] = *(const u32x4*)(Arow + koff + 16);
#pragma unroll
    for (int s = 0; s < 4; ++s) {
      Frag b;
      b.u[0] = *(const u32x4*)(Brow[s] + kk + kb);
      b.u[1] = *(const u32x4*)(Brow[s] + kk + kb + 16);
      acc[s] = __builtin_amdgcn_wmma_f32_16x16x32_bf16(false, a.v, false, b.v,
                                                       (short)0, acc[s], false, false);
    }
  }

  int rbase = wr * 16 + ((lane >= 16) ? 8 : 0);
#pragma unroll
  for (int s = 0; s < 4; ++s) {
    int col = wc * 64 + s * 16 + (lane & 15);
    float bcol = bp1[col];
#pragma unroll
    for (int r = 0; r < 8; ++r)
      hid[(rbase + r) * 260 + col] = acc[s][r] + bcol;
  }
  __syncthreads();

  int row = tid >> 3, c = tid & 7;
  float sum = bp2s[c];
  const float* h = hid + row * 260;
  const float* wcol = wp2s;                // wp2s layout: [k][c] flattened k*8+c
  for (int k = 0; k < 256; ++k) sum += h[k] * wcol[k * 8 + c];
  out[(e0 + row) * 8 + c] = sum;
}

// ---------------------------------------------------------------------------
// Host orchestration
// ---------------------------------------------------------------------------
static inline long cdiv(long a, long b) { return (a + b - 1) / b; }

extern "C" void kernel_launch(void* const* d_in, const int* in_sizes, int n_in,
                              void* d_out, int out_size, void* d_ws, size_t ws_size,
                              hipStream_t stream) {
  const float* x2   = (const float*)d_in[0];   // [N,256]
  const float* x3   = (const float*)d_in[1];   // [N,300]
  const int*   src  = (const int*)d_in[2];     // [R,E]
  const int*   dst  = (const int*)d_in[3];     // [R,E]
  const int*   dsrc = (const int*)d_in[4];     // [ED]
  const int*   ddst = (const int*)d_in[5];     // [ED]
  const float* W2a  = (const float*)d_in[6];
  const float* b2a  = (const float*)d_in[7];
  const float* W2b  = (const float*)d_in[8];
  const float* b2b  = (const float*)d_in[9];
  const float* W3a  = (const float*)d_in[10];
  const float* b3a  = (const float*)d_in[11];
  const float* W3b  = (const float*)d_in[12];
  const float* b3b  = (const float*)d_in[13];
  const float* Wp1  = (const float*)d_in[14];  // [512,256]
  const float* bp1  = (const float*)d_in[15];
  const float* Wp2  = (const float*)d_in[16];  // [256,8]
  const float* bp2  = (const float*)d_in[17];
  float* out = (float*)d_out;

  // bump allocator on workspace
  char* wsp = (char*)d_ws;
  auto alloc = [&](size_t bytes) -> void* {
    void* p = wsp;
    wsp += (bytes + 255) & ~(size_t)255;
    return p;
  };

  float*  norms = (float*)alloc((size_t)2 * R_ * N_ * 4);
  float*  ns = norms;
  float*  nd = norms + (long)R_ * N_;
  bf16_t* W2at  = (bf16_t*)alloc((size_t)R_ * 256 * 256 * 2);
  bf16_t* W2bt  = (bf16_t*)alloc((size_t)R_ * 128 * 256 * 2);
  bf16_t* W3at  = (bf16_t*)alloc((size_t)R_ * 256 * 320 * 2);
  bf16_t* W3bt  = (bf16_t*)alloc((size_t)R_ * 128 * 256 * 2);
  bf16_t* Wp1t  = (bf16_t*)alloc((size_t)256 * 512 * 2);
  bf16_t* x2b   = (bf16_t*)alloc((size_t)N_ * 256 * 2);
  bf16_t* x3b   = (bf16_t*)alloc((size_t)N_ * 320 * 2);
  bf16_t* hb    = (bf16_t*)alloc((size_t)N_ * 256 * 2);
  bf16_t* featb = (bf16_t*)alloc((size_t)N_ * 256 * 2);
  float*  tmp   = (float*)alloc((size_t)N_ * 256 * 4);
  float*  acc   = (float*)alloc((size_t)N_ * 256 * 4);

  // ---- degree norms (shared by both streams) ----
  {
    long n = (long)2 * R_ * N_;
    zero_f32<<<cdiv(n, 256), 256, 0, stream>>>(norms, n);
    long te = (long)R_ * E_;
    count_deg<<<cdiv(te, 256), 256, 0, stream>>>(src, ns, te, N_, E_);
    count_deg<<<cdiv(te, 256), 256, 0, stream>>>(dst, nd, te, N_, E_);
    norm_finalize<<<cdiv(n, 256), 256, 0, stream>>>(norms, n);
  }

  // ---- weight prep (transpose + bf16, K padded to /32) ----
  {
    long t;
    t = (long)R_ * 256 * 256;
    transpose_pad_bf16<<<cdiv(t, 256), 256, 0, stream>>>(W2a, W2at, 256, 256, 256, t);
    t = (long)R_ * 128 * 256;
    transpose_pad_bf16<<<cdiv(t, 256), 256, 0, stream>>>(W2b, W2bt, 256, 128, 256, t);
    t = (long)R_ * 256 * 320;
    transpose_pad_bf16<<<cdiv(t, 256), 256, 0, stream>>>(W3a, W3at, 300, 256, 320, t);
    t = (long)R_ * 128 * 256;
    transpose_pad_bf16<<<cdiv(t, 256), 256, 0, stream>>>(W3b, W3bt, 256, 128, 256, t);
    t = (long)256 * 512;
    transpose_pad_bf16<<<cdiv(t, 256), 256, 0, stream>>>(Wp1, Wp1t, 512, 256, 512, t);
  }

  // ---- feature conversion ----
  {
    long t = (long)N_ * 256;
    convert_pad_bf16<<<cdiv(t, 256), 256, 0, stream>>>(x2, x2b, 256, 256, t);
    t = (long)N_ * 320;
    convert_pad_bf16<<<cdiv(t, 256), 256, 0, stream>>>(x3, x3b, 300, 320, t);
  }

  // hetero_conv: acc = sum_r segsum( (x@W_r)[src]*ns_r[src]*nd_r[dst] ) ; + sum_r b_r
  auto conv = [&](const bf16_t* xb, int Kp, const bf16_t* Wt, int dout) {
    long na = (long)N_ * dout;
    zero_f32<<<cdiv(na, 256), 256, 0, stream>>>(acc, na);
    dim3 g((unsigned)cdiv(N_, 64), (unsigned)(dout / 64));
    for (int r = 0; r < R_; ++r) {
      gemm_bf16_wmma<<<g, 256, 0, stream>>>(xb, Wt + (long)r * dout * Kp, tmp, N_, Kp, dout);
      if (dout == 256)
        scatter_edges<256><<<cdiv(E_, 8), 256, 0, stream>>>(
            src + (long)r * E_, dst + (long)r * E_,
            ns + (long)r * N_, nd + (long)r * N_, tmp, acc, E_);
      else
        scatter_edges<128><<<cdiv(E_, 8), 256, 0, stream>>>(
            src + (long)r * E_, dst + (long)r * E_,
            ns + (long)r * N_, nd + (long)r * N_, tmp, acc, E_);
    }
  };

  // ---- stream 2 (node2_features) ----
  conv(x2b, 256, W2at, 256);
  finalize_relu_bf16<<<cdiv((long)N_ * 256, 256), 256, 0, stream>>>(acc, b2a, hb, (long)N_ * 256, 256);
  conv(hb, 256, W2bt, 128);
  finalize_feat_bf16<<<cdiv((long)N_ * 128, 256), 256, 0, stream>>>(acc, b2b, featb, (long)N_ * 128, 128, 256, 0);

  // ---- stream 3 (mpnn_features) ----
  conv(x3b, 320, W3at, 256);
  finalize_relu_bf16<<<cdiv((long)N_ * 256, 256), 256, 0, stream>>>(acc, b3a, hb, (long)N_ * 256, 256);
  conv(hb, 256, W3bt, 128);
  finalize_feat_bf16<<<cdiv((long)N_ * 128, 256), 256, 0, stream>>>(acc, b3b, featb, (long)N_ * 128, 128, 256, 128);

  // ---- fused decoder ----
  decoder_kernel<<<ED_ / 32, 256, 0, stream>>>(dsrc, ddst, featb, Wp1t, bp1, Wp2, bp2, out);
}
